// DoubleStream_Expert_30494267801903
// MI455X (gfx1250) — compile-verified, matrix operand order checked by reading, not run on hardware
//
#include <hip/hip_runtime.h>
#include <math.h>

// ---------------------------------------------------------------------------
// Types for CDNA5 WMMA (wave32)
// ---------------------------------------------------------------------------
typedef __bf16 bf16_t;
typedef __attribute__((ext_vector_type(16))) __bf16 v16bf;
typedef __attribute__((ext_vector_type(8)))  __bf16 v8bf;
typedef __attribute__((ext_vector_type(8)))  float  v8f;

#define WMMA_BF16(a, b, c) \
    __builtin_amdgcn_wmma_f32_16x16x32_bf16(false, (a), false, (b), (short)0, (c), false, false)

static __device__ inline v8f zero8() {
    v8f z = {0.f, 0.f, 0.f, 0.f, 0.f, 0.f, 0.f, 0.f};
    return z;
}

// ---------------------------------------------------------------------------
// Async global->LDS copy (16 bytes per lane), with sync fallback.
// Builtin signature (from hipcc diagnostic): param0 = int4 pointer in the
// global (__device__) address space; b128 => int4 pointee, non-const.
// ---------------------------------------------------------------------------
#if defined(__has_builtin) && __has_builtin(__builtin_amdgcn_global_load_async_to_lds_b128)
#define HAVE_ASYNC_LDS 1
#else
#define HAVE_ASYNC_LDS 0
#endif

typedef int v4i_vs __attribute__((vector_size(16)));
typedef __attribute__((address_space(1))) v4i_vs* as1_v4i;
typedef __attribute__((address_space(3))) v4i_vs* as3_v4i;

static __device__ inline void copy16_g2l(const bf16_t* g, bf16_t* l) {
#if HAVE_ASYNC_LDS
    __builtin_amdgcn_global_load_async_to_lds_b128(
        (as1_v4i)(void*)const_cast<bf16_t*>(g),
        (as3_v4i)(void*)l, 0, 0);
#else
    *(v8bf*)l = *(const v8bf*)g;
#endif
}

static __device__ inline void async_copies_wait() {
#if HAVE_ASYNC_LDS && __has_builtin(__builtin_amdgcn_s_wait_asynccnt)
    __builtin_amdgcn_s_wait_asynccnt(0);
#endif
}

// ---------------------------------------------------------------------------
// Fragment loads.  16x32 bf16 fragment (A layout; identical pattern serves B
// when the operand is stored [N][K] row-major).  Per ISA 7.12.2:
//   lanes 0-15 : row = lane,    K = 0..7  and 16..23
//   lanes 16-31: row = lane-16, K = 8..15 and 24..31
// ---------------------------------------------------------------------------
static __device__ inline v16bf load_frag(const bf16_t* __restrict__ p, int ld) {
    int lane = threadIdx.x & 31;
    const bf16_t* base = p + (size_t)(lane & 15) * ld + ((lane >> 4) << 3);
    v8bf lo = *(const v8bf*)(base);
    v8bf hi = *(const v8bf*)(base + 16);
    return __builtin_shufflevector(lo, hi, 0, 1, 2, 3, 4, 5, 6, 7,
                                   8, 9, 10, 11, 12, 13, 14, 15);
}

static __device__ inline v16bf load_frag_lds(const bf16_t* p, int ld) {
    int lane = threadIdx.x & 31;
    const bf16_t* base = p + (lane & 15) * ld + ((lane >> 4) << 3);
    v8bf lo = *(const v8bf*)(base);
    v8bf hi = *(const v8bf*)(base + 16);
    return __builtin_shufflevector(lo, hi, 0, 1, 2, 3, 4, 5, 6, 7,
                                   8, 9, 10, 11, 12, 13, 14, 15);
}

// ---------------------------------------------------------------------------
// Weight transpose + cast:  W[K][N] fp32  ->  Wt[N][K] bf16
// ---------------------------------------------------------------------------
__global__ void tcast_kernel(const float* __restrict__ w, bf16_t* __restrict__ wt,
                             int K, int N) {
    long idx = (long)blockIdx.x * 256 + threadIdx.x;
    if (idx < (long)K * N) {
        int k = (int)(idx / N);
        int n = (int)(idx % N);
        wt[(size_t)n * K + k] = (bf16_t)w[idx];
    }
}

// ---------------------------------------------------------------------------
// Modulation MLP (tiny: B=2)
// ---------------------------------------------------------------------------
__global__ void mod1_kernel(const float* __restrict__ p_emb,
                            const float* __restrict__ w1, const float* __restrict__ b1,
                            float* __restrict__ h1) {
    int b = blockIdx.x;          // 2
    int o = threadIdx.x;         // 512
    float acc = b1[o];
    for (int i = 0; i < 1024; ++i) {
        float x = p_emb[b * 1024 + i];
        float s = x / (1.f + __expf(-x));     // silu(p_emb)
        acc += s * w1[i * 512 + o];
    }
    h1[b * 512 + o] = acc / (1.f + __expf(-acc));  // silu
}

__global__ void mod2_kernel(const float* __restrict__ h1,
                            const float* __restrict__ w2, const float* __restrict__ b2,
                            float* __restrict__ modv) {
    int idx = blockIdx.x * 256 + threadIdx.x;   // 2*9216
    int b = idx / 9216, c = idx % 9216;
    float acc = b2[c];
    for (int o = 0; o < 512; ++o)
        acc += h1[b * 512 + o] * w2[o * 9216 + c];
    modv[idx] = acc;
}

// ---------------------------------------------------------------------------
// RMSNorm + modulation: out = bf16( rms(x)*w*(1+m_s) + m_h ), rows = 2048
// One wave per row (24 elements per lane).
// ---------------------------------------------------------------------------
__global__ __launch_bounds__(256) void rmsnorm_mod_kernel(
    const float* __restrict__ x, const float* __restrict__ w,
    const float* __restrict__ modv, int sc, int hc,
    bf16_t* __restrict__ out) {
    int wid = threadIdx.x >> 5, lane = threadIdx.x & 31;
    int row = blockIdx.x * 8 + wid;          // 2048 rows
    int b = row >> 10;
    const float* xr = x + (size_t)row * 768;
    float vals[24];
    float ss = 0.f;
#pragma unroll
    for (int e = 0; e < 24; ++e) {
        float v = xr[lane + e * 32];
        vals[e] = v;
        ss += v * v;
    }
#pragma unroll
    for (int m = 16; m >= 1; m >>= 1) ss += __shfl_xor(ss, m, 32);
    float r = rsqrtf(ss * (1.f / 768.f) + 1e-6f);
    const float* ms = modv + b * 9216 + sc * 768;
    const float* mh = modv + b * 9216 + hc * 768;
#pragma unroll
    for (int e = 0; e < 24; ++e) {
        int i = lane + e * 32;
        out[(size_t)row * 768 + i] = (bf16_t)(vals[e] * r * w[i] * (1.f + ms[i]) + mh[i]);
    }
}

// ---------------------------------------------------------------------------
// Generic bf16 WMMA GEMM with double-buffered async LDS staging.
//   C[M,N] = A[M,K] * Bt[N,K]^T + bias, epilogues:
//   MODE 0: out bf16 = C+bias                      (QKV)
//   MODE 1: out bf16 = gelu_exact(C+bias)          (MLP up)
//   MODE 2: outf = res + (C+bias)*gate  (M=4096 rows = [b][2T]; per-stream)
//   MODE 3: outf = res + (C+bias)*gate  (M=2048 rows, single stream)
// Block: 256 thr = 8 waves (2 in M x 4 in N); block tile 64x128; K-step 32.
// LDS: A 64x32, B 128x32, double buffered = 24 KB.
// ---------------------------------------------------------------------------
template <int MODE>
__global__ __launch_bounds__(256) void gemm_kernel(
    const bf16_t* __restrict__ A, const bf16_t* __restrict__ Bt,
    const float* __restrict__ bias,
    bf16_t* __restrict__ outb,
    float* __restrict__ outf, float* __restrict__ outf2,
    const float* __restrict__ res1, const float* __restrict__ res2,
    const float* __restrict__ g1, const float* __restrict__ g2,
    int M, int N, int K) {
    __shared__ __align__(16) bf16_t As[2][64 * 32];
    __shared__ __align__(16) bf16_t Bs[2][128 * 32];

    int tid = threadIdx.x;
    int wid = tid >> 5, lane = tid & 31;
    int mBlk = blockIdx.y * 64;
    int nBlk = blockIdx.x * 128;
    int mw = (wid & 1) * 32;     // wave offset within A tile
    int nw = (wid >> 1) * 32;    // wave offset within B tile

    // per-thread copy coordinates: 16B chunks
    int crow = tid >> 2;             // 0..63
    int cch = (tid & 3) * 8;         // 0,8,16,24 (bf16 elements)

    const bf16_t* gA = A + (size_t)(mBlk + crow) * K + cch;
    const bf16_t* gB0 = Bt + (size_t)(nBlk + crow) * K + cch;
    const bf16_t* gB1 = Bt + (size_t)(nBlk + crow + 64) * K + cch;
    bf16_t* lA = &As[0][crow * 32 + cch];
    bf16_t* lB0 = &Bs[0][crow * 32 + cch];
    bf16_t* lB1 = &Bs[0][(crow + 64) * 32 + cch];
    const int lbuf = 64 * 32;        // A buffer stride (elements)
    const int lbufB = 128 * 32;      // B buffer stride

    v8f acc[2][2];
    acc[0][0] = zero8(); acc[0][1] = zero8();
    acc[1][0] = zero8(); acc[1][1] = zero8();

    // prologue: stage k0 = 0 into buffer 0
    copy16_g2l(gA, lA);
    copy16_g2l(gB0, lB0);
    copy16_g2l(gB1, lB1);

    int buf = 0;
    for (int k0 = 0; k0 < K; k0 += 32, buf ^= 1) {
        async_copies_wait();
        __syncthreads();   // staged tile visible to all; previous buffer free
        if (k0 + 32 < K) {
            int nb = buf ^ 1;
            copy16_g2l(gA + k0 + 32, lA + nb * lbuf);
            copy16_g2l(gB0 + k0 + 32, lB0 + nb * lbufB);
            copy16_g2l(gB1 + k0 + 32, lB1 + nb * lbufB);
        }
        if (k0 + 64 < K) {
            __builtin_prefetch(gA + k0 + 64, 0, 1);
            __builtin_prefetch(gB0 + k0 + 64, 0, 1);
        }
        v16bf a0 = load_frag_lds(&As[buf][mw * 32], 32);
        v16bf a1 = load_frag_lds(&As[buf][(mw + 16) * 32], 32);
        v16bf b0 = load_frag_lds(&Bs[buf][nw * 32], 32);
        v16bf b1 = load_frag_lds(&Bs[buf][(nw + 16) * 32], 32);
        acc[0][0] = WMMA_BF16(a0, b0, acc[0][0]);
        acc[0][1] = WMMA_BF16(a0, b1, acc[0][1]);
        acc[1][0] = WMMA_BF16(a1, b0, acc[1][0]);
        acc[1][1] = WMMA_BF16(a1, b1, acc[1][1]);
    }

    int half = lane >> 4, nn = lane & 15;
#pragma unroll
    for (int i = 0; i < 2; ++i)
#pragma unroll
        for (int j = 0; j < 2; ++j)
#pragma unroll
            for (int r = 0; r < 8; ++r) {
                int row = mBlk + mw + i * 16 + r + half * 8;
                int col = nBlk + nw + j * 16 + nn;
                float v = acc[i][j][r] + bias[col];
                if (MODE == 0) {
                    outb[(size_t)row * N + col] = (bf16_t)v;
                } else if (MODE == 1) {
                    float g = 0.5f * v * (1.f + erff(v * 0.70710678f));
                    outb[(size_t)row * N + col] = (bf16_t)g;
                } else if (MODE == 2) {
                    int bb = row >> 11, nsq = row & 2047;
                    int s = nsq >> 10, t = nsq & 1023;
                    size_t ridx = ((size_t)(bb * 1024 + t)) * 768 + col;
                    const float* res = s ? res2 : res1;
                    const float* g = s ? g2 : g1;
                    float* of = s ? outf2 : outf;
                    of[ridx] = res[ridx] + v * g[bb * 9216 + col];
                } else {  // MODE 3
                    int bb = row >> 10;
                    size_t ridx = (size_t)row * 768 + col;
                    outf[ridx] = res1[ridx] + v * g1[bb * 9216 + col];
                }
            }
}

// ---------------------------------------------------------------------------
// RoPE + QK L2-norm + qk_scale, V transpose.  One wave per (b,h,n);
// lane owns dh = {lane, lane+32, lane+64}.  DH^-0.5 folded into q.
//   q,k out: [B][H][N][96] bf16 ;  vt out: [B][H][96][N] bf16
// ---------------------------------------------------------------------------
__global__ __launch_bounds__(128) void rope_kernel(
    const bf16_t* __restrict__ qkv1, const bf16_t* __restrict__ qkv2,
    const float* __restrict__ qk_scale,
    bf16_t* __restrict__ qo, bf16_t* __restrict__ ko, bf16_t* __restrict__ vt) {
    int wid = threadIdx.x >> 5, lane = threadIdx.x & 31;
    int task = blockIdx.x * 4 + wid;      // 32768 = B*H*N
    int n = task & 2047;
    int h = (task >> 11) & 7;
    int b = task >> 14;
    int s = n >> 10, t = n & 1023;
    const bf16_t* src = (s ? qkv2 : qkv1) + (size_t)(b * 1024 + t) * 2304;

    float qv[3], kv[3], vv[3];
    float ssq = 0.f, ssk = 0.f;
#pragma unroll
    for (int e = 0; e < 3; ++e) {
        int i = lane + e * 32;
        int j = (i < 48) ? i : i - 48;
        int p = (i < 48) ? i + 48 : i - 48;
        float sgn = (i < 48) ? -1.f : 1.f;
        // inv_freq rounded through bf16 as in the reference
        float invf = (float)(bf16_t)powf(10000.f, -(2.f * (float)j) / 96.f);
        float ang = (float)n * invf;
        float c = cosf(ang), sn = sinf(ang);
        float xq = (float)src[h * 96 + i];
        float pq = (float)src[h * 96 + p];
        float xk = (float)src[768 + h * 96 + i];
        float pk = (float)src[768 + h * 96 + p];
        qv[e] = xq * c + sgn * pq * sn;
        kv[e] = xk * c + sgn * pk * sn;
        vv[e] = (float)src[1536 + h * 96 + i];
        ssq += qv[e] * qv[e];
        ssk += kv[e] * kv[e];
    }
#pragma unroll
    for (int m = 16; m >= 1; m >>= 1) {
        ssq += __shfl_xor(ssq, m, 32);
        ssk += __shfl_xor(ssk, m, 32);
    }
    float iq = 1.f / fmaxf(sqrtf(ssq), 1e-12f);
    float ik = 1.f / fmaxf(sqrtf(ssk), 1e-12f);
    const float dh_isqrt = 0.10206207f;   // 96^-0.5, folded into q
    size_t base = ((size_t)(b * 8 + h) * 2048 + n) * 96;
#pragma unroll
    for (int e = 0; e < 3; ++e) {
        int i = lane + e * 32;
        float sc = qk_scale[i];
        qo[base + i] = (bf16_t)(qv[e] * iq * sc * dh_isqrt);
        ko[base + i] = (bf16_t)(kv[e] * ik * sc);
        vt[((size_t)(b * 8 + h) * 96 + i) * 2048 + n] = (bf16_t)vv[e];
    }
}

// ---------------------------------------------------------------------------
// Flash attention. One wave per (b,h, 16-query tile).  64 key-tiles of 32.
// scores: 6 WMMAs; online softmax (shfl within 16-lane halves); P tile goes
// through wave-private LDS (C-layout -> A-layout); PV: 6 WMMAs.
// attn out: [B][N][768] bf16
// ---------------------------------------------------------------------------
__global__ __launch_bounds__(256) void attn_kernel(
    const bf16_t* __restrict__ q, const bf16_t* __restrict__ k,
    const bf16_t* __restrict__ vt, bf16_t* __restrict__ attn) {
    __shared__ __align__(16) bf16_t pbuf[8][16 * 32];
    int wid = threadIdx.x >> 5, lane = threadIdx.x & 31;
    int task = blockIdx.x * 8 + wid;     // 2048 = B*H*(N/16)
    int qt = task & 127;
    int h = (task >> 7) & 7;
    int b = task >> 10;

    const bf16_t* qp = q + ((size_t)(b * 8 + h) * 2048 + qt * 16) * 96;
    const bf16_t* kp = k + (size_t)(b * 8 + h) * 2048 * 96;
    const bf16_t* vp = vt + (size_t)(b * 8 + h) * 96 * 2048;
    bf16_t* myp = pbuf[wid];

    v16bf qa[3];
#pragma unroll
    for (int d = 0; d < 3; ++d) qa[d] = load_frag(qp + d * 32, 96);

    v8f o[6];
#pragma unroll
    for (int tt = 0; tt < 6; ++tt) o[tt] = zero8();
    float mrow[8], lrow[8];
#pragma unroll
    for (int r = 0; r < 8; ++r) { mrow[r] = -1e30f; lrow[r] = 0.f; }

    int half = lane >> 4, nn = lane & 15;

    for (int j0 = 0; j0 < 2048; j0 += 32) {
        v8f s0 = zero8(), s1 = zero8();
#pragma unroll
        for (int d = 0; d < 3; ++d) {
            v16bf kb0 = load_frag(kp + (size_t)j0 * 96 + d * 32, 96);
            v16bf kb1 = load_frag(kp + (size_t)(j0 + 16) * 96 + d * 32, 96);
            s0 = WMMA_BF16(qa[d], kb0, s0);
            s1 = WMMA_BF16(qa[d], kb1, s1);
        }
        // online softmax over the 32 new columns
#pragma unroll
        for (int r = 0; r < 8; ++r) {
            float mx = fmaxf(s0[r], s1[r]);
#pragma unroll
            for (int m = 8; m >= 1; m >>= 1) mx = fmaxf(mx, __shfl_xor(mx, m, 16));
            float mnew = fmaxf(mrow[r], mx);
            float sc = __expf(mrow[r] - mnew);
            mrow[r] = mnew;
            float p0 = __expf(s0[r] - mnew);
            float p1 = __expf(s1[r] - mnew);
            float ps = p0 + p1;
#pragma unroll
            for (int m = 8; m >= 1; m >>= 1) ps += __shfl_xor(ps, m, 16);
            lrow[r] = lrow[r] * sc + ps;
#pragma unroll
            for (int tt = 0; tt < 6; ++tt) o[tt][r] *= sc;
            int row = r + half * 8;
            myp[row * 32 + nn] = (bf16_t)p0;
            myp[row * 32 + nn + 16] = (bf16_t)p1;
        }
        // reload P in A layout (wave-private region; DS ops in-order per wave)
        v16bf pa = load_frag_lds(myp, 32);
#pragma unroll
        for (int tt = 0; tt < 6; ++tt) {
            v16bf vb = load_frag(vp + (size_t)(tt * 16) * 2048 + j0, 2048);
            o[tt] = WMMA_BF16(pa, vb, o[tt]);
        }
    }

#pragma unroll
    for (int tt = 0; tt < 6; ++tt)
#pragma unroll
        for (int r = 0; r < 8; ++r) {
            int seqrow = qt * 16 + r + half * 8;
            int col = h * 96 + tt * 16 + nn;
            attn[((size_t)b * 2048 + seqrow) * 768 + col] = (bf16_t)(o[tt][r] / lrow[r]);
        }
}

// ---------------------------------------------------------------------------
// Host launcher
// ---------------------------------------------------------------------------
extern "C" void kernel_launch(void* const* d_in, const int* in_sizes, int n_in,
                              void* d_out, int out_size, void* d_ws, size_t ws_size,
                              hipStream_t stream) {
    (void)in_sizes; (void)n_in; (void)out_size; (void)ws_size;
    const float* x1      = (const float*)d_in[0];
    const float* x2      = (const float*)d_in[1];
    const float* p_emb   = (const float*)d_in[2];
    const float* mod_w1  = (const float*)d_in[3];
    const float* mod_b1  = (const float*)d_in[4];
    const float* mod_w2  = (const float*)d_in[5];
    const float* mod_b2  = (const float*)d_in[6];
    const float* norm11  = (const float*)d_in[7];
    const float* norm12  = (const float*)d_in[8];
    const float* qkv_w   = (const float*)d_in[9];
    const float* qkv_b   = (const float*)d_in[10];
    const float* qkv2_w  = (const float*)d_in[11];
    const float* qkv2_b  = (const float*)d_in[12];
    const float* qk_sc   = (const float*)d_in[13];
    const float* out_w   = (const float*)d_in[14];
    const float* out_b   = (const float*)d_in[15];
    const float* norm21  = (const float*)d_in[16];
    const float* norm22  = (const float*)d_in[17];
    const float* m1w1    = (const float*)d_in[18];
    const float* m1b1    = (const float*)d_in[19];
    const float* m1w2    = (const float*)d_in[20];
    const float* m1b2    = (const float*)d_in[21];
    const float* m2w1    = (const float*)d_in[22];
    const float* m2b1    = (const float*)d_in[23];
    const float* m2w2    = (const float*)d_in[24];
    const float* m2b2    = (const float*)d_in[25];

    float* outp  = (float*)d_out;
    float* out2p = outp + (size_t)2 * 1024 * 768;

    // ---- workspace arena ----
    char* ws = (char*)d_ws;
    size_t off = 0;
    auto alloc = [&](size_t bytes) -> void* {
        void* p = ws + off;
        off += (bytes + 255) & ~(size_t)255;
        return p;
    };
    bf16_t* wt_qkv1 = (bf16_t*)alloc((size_t)2304 * 768 * 2);
    bf16_t* wt_qkv2 = (bf16_t*)alloc((size_t)2304 * 768 * 2);
    bf16_t* wt_out  = (bf16_t*)alloc((size_t)768 * 768 * 2);
    bf16_t* wt_m11  = (bf16_t*)alloc((size_t)3072 * 768 * 2);
    bf16_t* wt_m12  = (bf16_t*)alloc((size_t)768 * 3072 * 2);
    bf16_t* wt_m21  = (bf16_t*)alloc((size_t)3072 * 768 * 2);
    bf16_t* wt_m22  = (bf16_t*)alloc((size_t)768 * 3072 * 2);
    float*  h1mod   = (float*)alloc((size_t)2 * 512 * 4);
    float*  modv    = (float*)alloc((size_t)2 * 9216 * 4);
    bf16_t* xm1     = (bf16_t*)alloc((size_t)2048 * 768 * 2);
    bf16_t* xm2     = (bf16_t*)alloc((size_t)2048 * 768 * 2);
    bf16_t* qkvo1   = (bf16_t*)alloc((size_t)2048 * 2304 * 2);
    bf16_t* qkvo2   = (bf16_t*)alloc((size_t)2048 * 2304 * 2);
    bf16_t* qbuf    = (bf16_t*)alloc((size_t)2 * 8 * 2048 * 96 * 2);
    bf16_t* kbuf    = (bf16_t*)alloc((size_t)2 * 8 * 2048 * 96 * 2);
    bf16_t* vtbuf   = (bf16_t*)alloc((size_t)2 * 8 * 96 * 2048 * 2);
    bf16_t* attnb   = (bf16_t*)alloc((size_t)2 * 2048 * 768 * 2);
    float*  x1buf   = (float*)alloc((size_t)2048 * 768 * 4);
    float*  x2buf   = (float*)alloc((size_t)2048 * 768 * 4);
    bf16_t* xn1     = (bf16_t*)alloc((size_t)2048 * 768 * 2);
    bf16_t* xn2     = (bf16_t*)alloc((size_t)2048 * 768 * 2);
    bf16_t* hh1     = (bf16_t*)alloc((size_t)2048 * 3072 * 2);
    bf16_t* hh2     = (bf16_t*)alloc((size_t)2048 * 3072 * 2);

    auto tcast = [&](const float* w, bf16_t* wt, int K, int N) {
        long total = (long)K * N;
        tcast_kernel<<<(unsigned)((total + 255) / 256), 256, 0, stream>>>(w, wt, K, N);
    };

    // 1) weights -> [N][K] bf16
    tcast(qkv_w, wt_qkv1, 768, 2304);
    tcast(qkv2_w, wt_qkv2, 768, 2304);
    tcast(out_w, wt_out, 768, 768);
    tcast(m1w1, wt_m11, 768, 3072);
    tcast(m1w2, wt_m12, 3072, 768);
    tcast(m2w1, wt_m21, 768, 3072);
    tcast(m2w2, wt_m22, 3072, 768);

    // 2) modulation MLP
    mod1_kernel<<<2, 512, 0, stream>>>(p_emb, mod_w1, mod_b1, h1mod);
    mod2_kernel<<<72, 256, 0, stream>>>(h1mod, mod_w2, mod_b2, modv);

    // 3) pre-attention norms (chunks: 0 m1s,1 m1h | 3 m2s,4 m2h)
    rmsnorm_mod_kernel<<<256, 256, 0, stream>>>(x1, norm11, modv, 0, 1, xm1);
    rmsnorm_mod_kernel<<<256, 256, 0, stream>>>(x2, norm12, modv, 3, 4, xm2);

    // 4) QKV GEMMs (M=2048, N=2304, K=768)
    gemm_kernel<0><<<dim3(18, 32), 256, 0, stream>>>(
        xm1, wt_qkv1, qkv_b, qkvo1, nullptr, nullptr, nullptr, nullptr,
        nullptr, nullptr, 2048, 2304, 768);
    gemm_kernel<0><<<dim3(18, 32), 256, 0, stream>>>(
        xm2, wt_qkv2, qkv2_b, qkvo2, nullptr, nullptr, nullptr, nullptr,
        nullptr, nullptr, 2048, 2304, 768);

    // 5) RoPE + L2 norm + scale, V transpose
    rope_kernel<<<8192, 128, 0, stream>>>(qkvo1, qkvo2, qk_sc, qbuf, kbuf, vtbuf);

    // 6) attention
    attn_kernel<<<256, 256, 0, stream>>>(qbuf, kbuf, vtbuf, attnb);

    // 7) out projection + gated residual (M=4096, N=768, K=768)
    gemm_kernel<2><<<dim3(6, 64), 256, 0, stream>>>(
        attnb, wt_out, out_b, nullptr, x1buf, x2buf, x1, x2,
        modv + 2 * 768, modv + 5 * 768, 4096, 768, 768);

    // 8) post-attention norms (chunks: 6 m3s,7 m3h | 9 m4s,10 m4h)
    rmsnorm_mod_kernel<<<256, 256, 0, stream>>>(x1buf, norm21, modv, 6, 7, xn1);
    rmsnorm_mod_kernel<<<256, 256, 0, stream>>>(x2buf, norm22, modv, 9, 10, xn2);

    // 9) MLP up + exact GELU (M=2048, N=3072, K=768)
    gemm_kernel<1><<<dim3(24, 32), 256, 0, stream>>>(
        xn1, wt_m11, m1b1, hh1, nullptr, nullptr, nullptr, nullptr,
        nullptr, nullptr, 2048, 3072, 768);
    gemm_kernel<1><<<dim3(24, 32), 256, 0, stream>>>(
        xn2, wt_m21, m2b1, hh2, nullptr, nullptr, nullptr, nullptr,
        nullptr, nullptr, 2048, 3072, 768);

    // 10) MLP down + gated residual -> final outputs (M=2048, N=768, K=3072)
    gemm_kernel<3><<<dim3(6, 32), 256, 0, stream>>>(
        hh1, wt_m12, m1b2, nullptr, outp, nullptr, x1buf, nullptr,
        modv + 8 * 768, nullptr, 2048, 768, 3072);
    gemm_kernel<3><<<dim3(6, 32), 256, 0, stream>>>(
        hh2, wt_m22, m2b2, nullptr, out2p, nullptr, x2buf, nullptr,
        modv + 11 * 768, nullptr, 2048, 768, 3072);
}
